// EdgeBlock_31885837206099
// MI455X (gfx1250) — matrix operand order repaired and checked by reading, not compiled, not run on hardware
//
#include <hip/hip_runtime.h>

#define D_E    128
#define D_N    128
#define N_OUT  128
#define K_TOT  384
#define M_TILE 128
#define KSTEPS 12          // 384 / 32
#define LDS_KPAD 40        // 32 k-values + 8 pad (bf16 units) -> bank-friendly stride

typedef __attribute__((ext_vector_type(16))) __bf16        v16bf;
typedef __attribute__((ext_vector_type(8)))  float         v8f;
typedef __attribute__((ext_vector_type(4)))  float         f32x4;
typedef __attribute__((ext_vector_type(4)))  unsigned int  u32x4;

union V16 {
    v16bf          v;
    unsigned short u[16];
    u32x4          q2[2];
};

// Native conversions: let clang emit the hardware bf16 cvt path (RNE).
__device__ __forceinline__ __bf16 f2bf(float f) { return (__bf16)f; }
__device__ __forceinline__ float  bf2f(__bf16 h) { return (float)h; }
__device__ __forceinline__ unsigned short bfbits(__bf16 h) {
    union { __bf16 b; unsigned short s; } c; c.b = h; return c.s;
}

__global__ __launch_bounds__(256) void edgeblock_wmma_kernel(
    const float* __restrict__ edge_feat,
    const float* __restrict__ node_feat,
    const int*   __restrict__ src_idx,
    const int*   __restrict__ dst_idx,
    const float* __restrict__ W,
    const float* __restrict__ bias,
    float*       __restrict__ out,
    int n_edges)
{
    // Double-buffered bf16 hi/lo W chunks: 2 bufs * 2 mats * 128n * 40k * 2B = 40 KB
    __shared__ unsigned short sWhi[2][N_OUT * LDS_KPAD];
    __shared__ unsigned short sWlo[2][N_OUT * LDS_KPAD];

    const int tid  = threadIdx.x;
    const int lane = tid & 31;
    const int wv   = tid >> 5;       // wave 0..7, each owns 16 rows
    const int l15  = lane & 15;
    const int half = lane >> 4;

    // Stage K-chunk kc (32 k-values x 128 n) of W into LDS buffer bb as bf16 hi/lo.
    auto stage = [&](int kc, int bb) {
        const float* Wk = W + (size_t)kc * 32 * N_OUT;
        #pragma unroll
        for (int i = 0; i < (32 * N_OUT) / 256; ++i) {
            int idx = tid + i * 256;
            int kk  = idx >> 7;        // 0..31
            int n   = idx & 127;       // coalesced over n
            float f = Wk[kk * N_OUT + n];
            __bf16 h = f2bf(f);
            __bf16 l = f2bf(f - bf2f(h));
            sWhi[bb][n * LDS_KPAD + kk] = bfbits(h);   // n-major: k contiguous per n
            sWlo[bb][n * LDS_KPAD + kk] = bfbits(l);
        }
    };

    stage(0, 0);

    // Per-lane row (A-layout: lanes 0-15 and 16-31 both hold rows M=0..15)
    int e = blockIdx.x * M_TILE + wv * 16 + l15;
    if (e >= n_edges) e = n_edges - 1;           // clamp: keeps EXEC all-1s
    const int si = src_idx[e];
    const int di = dst_idx[e];
    const float* pe = edge_feat + (size_t)e  * D_E;
    const float* ps = node_feat + (size_t)si * D_N;
    const float* pd = node_feat + (size_t)di * D_N;

    // Accumulators init with bias (C/D layout: lane -> N = lane%16)
    v8f acc[8];
    #pragma unroll
    for (int nt = 0; nt < 8; ++nt) {
        float bv = bias[nt * 16 + l15];
        #pragma unroll
        for (int r = 0; r < 8; ++r) acc[nt][r] = bv;
    }

    __syncthreads();

    for (int ks = 0; ks < KSTEPS; ++ks) {
        const int bb = ks & 1;
        if (ks + 1 < KSTEPS) stage(ks + 1, bb ^ 1);   // prefetch next W chunk

        // ---- A fragment: 16x32 bf16, rows = per-lane e, K = [32ks, 32ks+32) ----
        // A layout: elems 0-7 <-> K = 8*half+0..7 ; elems 8-15 <-> K = 16+8*half+0..7
        const int reg  = ks >> 2;              // 0: edge, 1: src node, 2: dst node
        const int koff = (ks & 3) << 5;        // k offset within the 128-wide region
        const float* bp = (reg == 0) ? pe : ((reg == 1) ? ps : pd);
        const float* p  = bp + koff + 8 * half;

        f32x4 f0, f1, f2, f3;
        if (reg == 0) {  // streaming edge_feat: non-temporal, protect L2 for node_feat
            f0 = __builtin_nontemporal_load((const f32x4*)(p + 0));
            f1 = __builtin_nontemporal_load((const f32x4*)(p + 4));
            f2 = __builtin_nontemporal_load((const f32x4*)(p + 16));
            f3 = __builtin_nontemporal_load((const f32x4*)(p + 20));
        } else {         // gathered node rows: want L2 residency
            f0 = *(const f32x4*)(p + 0);
            f1 = *(const f32x4*)(p + 4);
            f2 = *(const f32x4*)(p + 16);
            f3 = *(const f32x4*)(p + 20);
        }

        float av[16] = { f0.x, f0.y, f0.z, f0.w,  f1.x, f1.y, f1.z, f1.w,
                         f2.x, f2.y, f2.z, f2.w,  f3.x, f3.y, f3.z, f3.w };
        V16 Ah, Al;
        #pragma unroll
        for (int i = 0; i < 16; ++i) {
            __bf16 h = f2bf(av[i]);
            __bf16 l = f2bf(av[i] - bf2f(h));   // fp32-grade split: a = ah + al
            Ah.v[i] = h;
            Al.v[i] = l;
        }

        // ---- B fragments from LDS + 3-term WMMA per 16-wide N tile ----
        // B (32x16) layout: lane -> N = lane%16 ; elems 0-7 <-> K = 8*half+0..7 ; elems 8-15 <-> K = 16+8*half+0..7
        #pragma unroll
        for (int nt = 0; nt < 8; ++nt) {
            const int off = (nt * 16 + l15) * LDS_KPAD + 8 * half;
            V16 Bh, Bl;
            Bh.q2[0] = *(const u32x4*)(&sWhi[bb][off]);
            Bh.q2[1] = *(const u32x4*)(&sWhi[bb][off + 16]);
            Bl.q2[0] = *(const u32x4*)(&sWlo[bb][off]);
            Bl.q2[1] = *(const u32x4*)(&sWlo[bb][off + 16]);

            acc[nt] = __builtin_amdgcn_wmma_f32_16x16x32_bf16(
                false, Ah.v, false, Bh.v, (short)0, acc[nt], false, false);
            acc[nt] = __builtin_amdgcn_wmma_f32_16x16x32_bf16(
                false, Al.v, false, Bh.v, (short)0, acc[nt], false, false);
            acc[nt] = __builtin_amdgcn_wmma_f32_16x16x32_bf16(
                false, Ah.v, false, Bl.v, (short)0, acc[nt], false, false);
        }

        __syncthreads();
    }

    // ---- Epilogue: C/D layout lane -> N = lane%16, VGPR r -> M = r + 8*half ----
    const int rbase = blockIdx.x * M_TILE + wv * 16 + 8 * half;
    #pragma unroll
    for (int r = 0; r < 8; ++r) {
        const int row = rbase + r;
        if (row < n_edges) {
            float* orow = out + (size_t)row * N_OUT + l15;
            #pragma unroll
            for (int nt = 0; nt < 8; ++nt)
                __builtin_nontemporal_store(acc[nt][r], orow + nt * 16);
        }
    }
}

extern "C" void kernel_launch(void* const* d_in, const int* in_sizes, int n_in,
                              void* d_out, int out_size, void* d_ws, size_t ws_size,
                              hipStream_t stream) {
    const float* edge_feat = (const float*)d_in[0];
    const float* node_feat = (const float*)d_in[1];
    const int*   src_idx   = (const int*)d_in[2];
    const int*   dst_idx   = (const int*)d_in[3];
    const float* W         = (const float*)d_in[4];
    const float* b         = (const float*)d_in[5];
    float* out = (float*)d_out;

    const int n_edges = in_sizes[2];
    const int blocks  = (n_edges + M_TILE - 1) / M_TILE;

    edgeblock_wmma_kernel<<<blocks, 256, 0, stream>>>(
        edge_feat, node_feat, src_idx, dst_idx, W, b, out, n_edges);
}
